// MultiHeadAttention_25323127177939
// MI455X (gfx1250) — matrix operand order repaired and checked by reading, not compiled, not run on hardware
//
#include <hip/hip_runtime.h>
#include <hip/hip_bf16.h>

typedef _Float16 v16h __attribute__((ext_vector_type(16)));
typedef _Float16 v8h  __attribute__((ext_vector_type(8)));
typedef float    v8f  __attribute__((ext_vector_type(8)));
typedef int      v4i  __attribute__((ext_vector_type(4)));

#define NB  4
#define NS  2048
#define ND  1024
#define NH  16
#define NHS 64

#if defined(__has_builtin)
#if __has_builtin(__builtin_amdgcn_global_load_async_to_lds_b128)
#define USE_ASYNC_LDS 1
#endif
#endif

static __device__ __forceinline__ v8f wmma_f16(v16h a, v16h b, v8f c) {
  // D = A(16x32 f16) * B(32x16 f16) + C(16x16 f32)
  return __builtin_amdgcn_wmma_f32_16x16x32_f16(
      /*neg_a=*/false, a, /*neg_b=*/false, b,
      /*c_mod=*/(short)0, c, /*reuse_a=*/false, /*reuse_b=*/false);
}

static __device__ __forceinline__ v16h cat8(v8h lo, v8h hi) {
  return __builtin_shufflevector(lo, hi, 0, 1, 2, 3, 4, 5, 6, 7,
                                 8, 9, 10, 11, 12, 13, 14, 15);
}

// 16-byte global -> LDS copy; async (ASYNCcnt) when the toolchain exposes it.
static __device__ __forceinline__ void copy16_g2l(const _Float16* g,
                                                  _Float16* l) {
#ifdef USE_ASYNC_LDS
  __builtin_amdgcn_global_load_async_to_lds_b128(
      (__attribute__((address_space(1))) v4i*)g,
      (__attribute__((address_space(3))) v4i*)l, 0, 0);
#else
  *(v8h*)l = *(const v8h*)g;
#endif
}

static __device__ __forceinline__ void wait_g2l() {
#ifdef USE_ASYNC_LDS
#if __has_builtin(__builtin_amdgcn_s_wait_asynccnt)
  __builtin_amdgcn_s_wait_asynccnt(0);
#else
  asm volatile("s_wait_asynccnt 0x0" ::: "memory");
#endif
#endif
}

// ---------------------------------------------------------------------------
// Kernel 1: fused per-head QKV projection.  grid = (S/128, B*H), block = 256.
// Q,K stored f16 [bh, S, HS]; V stored TRANSPOSED f16 [bh, HS, S].
// ---------------------------------------------------------------------------
__global__ __launch_bounds__(256) void qkv_proj_kernel(
    const float* __restrict__ X, const float* __restrict__ Wq,
    const float* __restrict__ Wk, const float* __restrict__ Wv,
    _Float16* __restrict__ Qh, _Float16* __restrict__ Kh,
    _Float16* __restrict__ Vt) {
  __shared__ alignas(32) _Float16 xs[128 * 32];   // X tile, row-major [row][k]
  __shared__ alignas(32) _Float16 wqs[64 * 32];   // W tiles transposed [e][k]
  __shared__ alignas(32) _Float16 wks[64 * 32];
  __shared__ alignas(32) _Float16 wvs[64 * 32];

  const int tid  = threadIdx.x;
  const int wave = tid >> 5;
  const int lane = tid & 31;
  const int ln   = lane & 15;   // N / M-within-16
  const int hh   = lane >> 4;   // lane half selects K sub-range

  const int s0 = blockIdx.x * 128;
  const int bh = blockIdx.y;
  const int b  = bh >> 4;
  const int h  = bh & 15;

  const float* Xb  = X  + (size_t)b * NS * ND;
  const float* Wqh = Wq + (size_t)h * ND * NHS;
  const float* Wkh = Wk + (size_t)h * ND * NHS;
  const float* Wvh = Wv + (size_t)h * ND * NHS;

  v8f zero = {};
  v8f aq[4], ak[4], av[4];
#pragma unroll
  for (int n = 0; n < 4; ++n) { aq[n] = zero; ak[n] = zero; av[n] = zero; }

  for (int kk = 0; kk < ND; kk += 32) {
    __syncthreads();
    // stage X tile (128 x 32) fp32 -> f16, coalesced along k
#pragma unroll
    for (int i = tid; i < 128 * 32; i += 256) {
      int r = i >> 5, c = i & 31;
      xs[i] = (_Float16)Xb[(size_t)(s0 + r) * ND + kk + c];
    }
    // stage W tiles (32 x 64) transposed into [e][k], coalesced along e
#pragma unroll
    for (int i = tid; i < 64 * 32; i += 256) {
      int k = i >> 6, e = i & 63;
      size_t g  = (size_t)(kk + k) * NHS + e;
      int    li = e * 32 + k;
      wqs[li] = (_Float16)Wqh[g];
      wks[li] = (_Float16)Wkh[g];
      wvs[li] = (_Float16)Wvh[g];
    }
    if (kk + 32 < ND)
      __builtin_prefetch(&Xb[(size_t)(s0 + (tid >> 5)) * ND + kk + 32], 0, 1);
    __syncthreads();

    // A fragment: lane M = ln, K = {hh*16..hh*16+15}
    const int row = wave * 16 + ln;
    v8h  alo = *(const v8h*)&xs[row * 32 + hh * 16];
    v8h  ahi = *(const v8h*)&xs[row * 32 + hh * 16 + 8];
    v16h a   = cat8(alo, ahi);

    // group loads before WMMAs so waits batch per matrix
    v16h bf[4];
#pragma unroll
    for (int n = 0; n < 4; ++n)
      bf[n] = *(const v16h*)&wqs[(n * 16 + ln) * 32 + hh * 16];
#pragma unroll
    for (int n = 0; n < 4; ++n) aq[n] = wmma_f16(a, bf[n], aq[n]);
#pragma unroll
    for (int n = 0; n < 4; ++n)
      bf[n] = *(const v16h*)&wks[(n * 16 + ln) * 32 + hh * 16];
#pragma unroll
    for (int n = 0; n < 4; ++n) ak[n] = wmma_f16(a, bf[n], ak[n]);
#pragma unroll
    for (int n = 0; n < 4; ++n)
      bf[n] = *(const v16h*)&wvs[(n * 16 + ln) * 32 + hh * 16];
#pragma unroll
    for (int n = 0; n < 4; ++n) av[n] = wmma_f16(a, bf[n], av[n]);
  }

  // C/D layout: lane holds N = ln, rows M = r (+8 for upper half)
#pragma unroll
  for (int n = 0; n < 4; ++n)
#pragma unroll
    for (int r = 0; r < 8; ++r) {
      int m    = hh ? r + 8 : r;
      int srow = s0 + wave * 16 + m;
      int e    = n * 16 + ln;
      Qh[((size_t)bh * NS + srow) * NHS + e] = (_Float16)aq[n][r];
      Kh[((size_t)bh * NS + srow) * NHS + e] = (_Float16)ak[n][r];
      Vt[((size_t)bh * NHS + e) * NS + srow] = (_Float16)av[n][r];
    }
}

// ---------------------------------------------------------------------------
// Kernel 2: causal flash attention.  grid = (S/128, B*H), block = 256.
// Block-wide key loop; K/V tiles staged once per block via async LDS copies
// and shared by all 8 waves.  Each wave owns 16 query rows.
// ---------------------------------------------------------------------------
__global__ __launch_bounds__(256) void flash_attn_kernel(
    const _Float16* __restrict__ Qh, const _Float16* __restrict__ Kh,
    const _Float16* __restrict__ Vt, _Float16* __restrict__ Oc) {
  __shared__ alignas(32) _Float16 ks[32 * 64];     // K tile  [key][e]
  __shared__ alignas(32) _Float16 vs[64 * 32];     // V tile  [e][key]
  __shared__ alignas(32) _Float16 ps[8][16 * 32];  // per-wave P staging

  const int tid  = threadIdx.x;
  const int wave = tid >> 5;
  const int lane = tid & 31;
  const int ln   = lane & 15;
  const int hh   = lane >> 4;

  const int bh  = blockIdx.y;
  const int b   = bh >> 4;
  const int h   = bh & 15;
  const int q0b = blockIdx.x * 128;
  const int q0w = q0b + wave * 16;

  const _Float16* Qbh = Qh + (size_t)bh * NS * NHS;
  const _Float16* Kbh = Kh + (size_t)bh * NS * NHS;
  const _Float16* Vbh = Vt + (size_t)bh * NHS * NS;

  // Q A-fragments: 2 chunks of K=32 across HS=64
  v16h qa[2];
  {
    const _Float16* qp = &Qbh[(size_t)(q0w + ln) * NHS];
#pragma unroll
    for (int c = 0; c < 2; ++c) {
      v8h lo = *(const v8h*)(qp + c * 32 + hh * 16);
      v8h hi = *(const v8h*)(qp + c * 32 + hh * 16 + 8);
      qa[c]  = cat8(lo, hi);
    }
  }

  v8f zero = {};
  v8f oacc[4];
#pragma unroll
  for (int n = 0; n < 4; ++n) oacc[n] = zero;
  float m_r[8], l_r[8];
#pragma unroll
  for (int r = 0; r < 8; ++r) { m_r[r] = -1e30f; l_r[r] = 0.0f; }

  const int kendb = q0b + 128;  // block causal bound (<= NS, so no OOB)
  for (int k0 = 0; k0 < kendb; k0 += 32) {
    __syncthreads();  // previous-iteration tile reads done
    // stage K tile: 32 keys x 64 f16 (4 KB); 256 lanes x 16 B
    {
      const int krow = tid >> 3;          // 0..31
      const int kc   = (tid & 7) * 8;     // f16 offset, 16B chunk
      copy16_g2l(&Kbh[(size_t)(k0 + krow) * NHS + kc], &ks[krow * 64 + kc]);
      // stage V tile: 64 e-rows x 32 keys f16 (4 KB)
      const int e  = tid >> 2;            // 0..63
      const int vc = (tid & 3) * 8;
      copy16_g2l(&Vbh[(size_t)e * NS + k0 + vc], &vs[e * 32 + vc]);
    }
    wait_g2l();
    __syncthreads();

    // ---- S = Q @ K^T from LDS (B-fragment: per-lane contiguous HS chunk)
    v8f sacc[2];
    sacc[0] = zero; sacc[1] = zero;
#pragma unroll
    for (int t = 0; t < 2; ++t) {
#pragma unroll
      for (int c = 0; c < 2; ++c) {
        v16h kb = *(const v16h*)&ks[(t * 16 + ln) * 64 + c * 32 + hh * 16];
        sacc[t] = wmma_f16(qa[c], kb, sacc[t]);
      }
    }

    // ---- online softmax per query row (16 lanes of one half-wave)
#pragma unroll
    for (int r = 0; r < 8; ++r) {
      const int M  = hh ? r + 8 : r;
      const int qg = q0w + M;
      float s0 = sacc[0][r] * 0.125f;  // 1/sqrt(64)
      float s1 = sacc[1][r] * 0.125f;
      if (k0 + ln      > qg) s0 = -1e30f;  // causal mask
      if (k0 + 16 + ln > qg) s1 = -1e30f;
      float mx = fmaxf(s0, s1);
#pragma unroll
      for (int off = 8; off > 0; off >>= 1)
        mx = fmaxf(mx, __shfl_xor(mx, off, 32));
      mx = fmaxf(mx, m_r[r]);
      float alpha = __expf(m_r[r] - mx);
      float p0 = __expf(s0 - mx);
      float p1 = __expf(s1 - mx);
      float rs = p0 + p1;
#pragma unroll
      for (int off = 8; off > 0; off >>= 1)
        rs += __shfl_xor(rs, off, 32);
      m_r[r] = mx;
      l_r[r] = l_r[r] * alpha + rs;
#pragma unroll
      for (int n = 0; n < 4; ++n) oacc[n][r] *= alpha;
      // stage P (score layout) for relayout to A-fragment layout
      ps[wave][M * 32 + ln]      = (_Float16)p0;
      ps[wave][M * 32 + 16 + ln] = (_Float16)p1;
    }
    asm volatile("" ::: "memory");  // keep ds stores before ds loads
    // ---- O += P @ V (A-fragment of P via wave-local LDS; LDS is in-order)
    v8h  plo = *(const v8h*)&ps[wave][ln * 32 + hh * 16];
    v8h  phi = *(const v8h*)&ps[wave][ln * 32 + hh * 16 + 8];
    v16h pa  = cat8(plo, phi);
    v16h vf[4];
#pragma unroll
    for (int n = 0; n < 4; ++n)
      vf[n] = *(const v16h*)&vs[(n * 16 + ln) * 32 + hh * 16];
#pragma unroll
    for (int n = 0; n < 4; ++n) oacc[n] = wmma_f16(pa, vf[n], oacc[n]);
  }

  // epilogue: normalize by l, store concat-head f16 [b, s, h*HS + e]
#pragma unroll
  for (int r = 0; r < 8; ++r) {
    const int   M    = hh ? r + 8 : r;
    const int   srow = q0w + M;
    const float inv  = 1.0f / l_r[r];
#pragma unroll
    for (int n = 0; n < 4; ++n) {
      Oc[((size_t)(b * NS + srow)) * ND + h * NHS + n * 16 + ln] =
          (_Float16)(oacc[n][r] * inv);
    }
  }
}

// ---------------------------------------------------------------------------
// Kernel 3: output projection Y = O @ Wo + bo.
// grid = (B*S/128, D/64), block = 256.  O tile async-staged to LDS.
// ---------------------------------------------------------------------------
__global__ __launch_bounds__(256) void out_proj_kernel(
    const _Float16* __restrict__ Oc, const float* __restrict__ Wo,
    const float* __restrict__ bo, float* __restrict__ Y) {
  __shared__ alignas(32) _Float16 wos[64 * 32];   // Wo tile transposed [e][k]
  __shared__ alignas(32) _Float16 osb[128 * 32];  // O tile [row][k]

  const int tid  = threadIdx.x;
  const int wave = tid >> 5;
  const int lane = tid & 31;
  const int ln   = lane & 15;
  const int hh   = lane >> 4;

  const int row0 = blockIdx.x * 128;
  const int c0   = blockIdx.y * 64;

  v8f zero = {};
  v8f acc[4];
#pragma unroll
  for (int n = 0; n < 4; ++n) acc[n] = zero;

  for (int kk = 0; kk < ND; kk += 32) {
    __syncthreads();
    // async stage O tile: 128 rows x 32 f16 (8 KB) = 512 chunks of 16 B
#pragma unroll
    for (int i = tid; i < 128 * 4; i += 256) {
      int row = i >> 2, cb = (i & 3) * 8;
      copy16_g2l(&Oc[(size_t)(row0 + row) * ND + kk + cb],
                 &osb[row * 32 + cb]);
    }
    // stage Wo tile transposed [e][k] with fp32->f16 convert
#pragma unroll
    for (int i = tid; i < 64 * 32; i += 256) {
      int k = i >> 6, e = i & 63;
      wos[e * 32 + k] = (_Float16)Wo[(size_t)(kk + k) * ND + c0 + e];
    }
    if (kk + 32 < ND)
      __builtin_prefetch(&Wo[(size_t)(kk + 32 + (tid >> 6)) * ND + c0], 0, 1);
    wait_g2l();
    __syncthreads();

    v8h  alo = *(const v8h*)&osb[(wave * 16 + ln) * 32 + hh * 16];
    v8h  ahi = *(const v8h*)&osb[(wave * 16 + ln) * 32 + hh * 16 + 8];
    v16h a   = cat8(alo, ahi);
    v16h bf[4];
#pragma unroll
    for (int n = 0; n < 4; ++n)
      bf[n] = *(const v16h*)&wos[(n * 16 + ln) * 32 + hh * 16];
#pragma unroll
    for (int n = 0; n < 4; ++n) acc[n] = wmma_f16(a, bf[n], acc[n]);
  }

#pragma unroll
  for (int n = 0; n < 4; ++n)
#pragma unroll
    for (int r = 0; r < 8; ++r) {
      int m   = hh ? r + 8 : r;
      int row = row0 + wave * 16 + m;
      int col = c0 + n * 16 + ln;
      Y[(size_t)row * ND + col] = acc[n][r] + bo[col];
    }
}

// ---------------------------------------------------------------------------
extern "C" void kernel_launch(void* const* d_in, const int* in_sizes, int n_in,
                              void* d_out, int out_size, void* d_ws,
                              size_t ws_size, hipStream_t stream) {
  (void)in_sizes; (void)n_in; (void)out_size; (void)ws_size;
  const float* X  = (const float*)d_in[0];
  const float* Wq = (const float*)d_in[1];
  const float* Wk = (const float*)d_in[2];
  const float* Wv = (const float*)d_in[3];
  const float* Wo = (const float*)d_in[4];
  const float* bo = (const float*)d_in[5];
  float* Y = (float*)d_out;

  const size_t elems = (size_t)NB * NH * NS * NHS;       // 8M f16 per tensor
  const size_t seg   = elems * sizeof(_Float16) + 4096;  // padded segments
  char* ws = (char*)d_ws;
  _Float16* Qh = (_Float16*)(ws + 0 * seg);
  _Float16* Kh = (_Float16*)(ws + 1 * seg);
  _Float16* Vt = (_Float16*)(ws + 2 * seg);
  _Float16* Oc = (_Float16*)(ws + 3 * seg);

  qkv_proj_kernel<<<dim3(NS / 128, NB * NH), 256, 0, stream>>>(
      X, Wq, Wk, Wv, Qh, Kh, Vt);
  flash_attn_kernel<<<dim3(NS / 128, NB * NH), 256, 0, stream>>>(
      Qh, Kh, Vt, Oc);
  out_proj_kernel<<<dim3((NB * NS) / 128, ND / 64), 256, 0, stream>>>(
      Oc, Wo, bo, Y);
}
